// ParallelExpertsMixer_25039659336212
// MI455X (gfx1250) — compile-verified, matrix operand order checked by reading, not run on hardware
//
#include <hip/hip_runtime.h>
#include <hip/hip_bf16.h>
#include <math.h>

typedef __bf16 bf16;
typedef __attribute__((ext_vector_type(16))) __bf16 v16bf;
typedef __attribute__((ext_vector_type(8)))  __bf16 bf16x8;
typedef __attribute__((ext_vector_type(8)))  float  v8f;

#define WAVES  8
#define LDA    520      // bf16 row stride for 512-wide buffers (pad -> conflict-free, 16B aligned)
#define LDS2   1032     // bf16 row stride for 1024-wide buffer
#define LDS2F  516      // f32 view stride of s2
#define TILE_S 16       // samples per workgroup
#define ROWS   64       // token rows per workgroup (= TILE_S * 4 branches)

// packed bf16 weight offsets inside d_ws (elements)
#define OFF_CONVPW  0u
#define OFF_INPROJ  262144u
#define OFF_ATTNOUT 1048576u
#define OFF_FFN1    1310720u
#define OFF_FFN2    1835008u
#define OFF_MLP1    2359296u
#define OFF_MLP2    2883584u
#define OFF_GATE1   3407872u
#define OFF_SE1     3538944u
#define OFF_SE2     3571712u
#define OFF_OUTP    3604480u

// ---------------------------------------------------------------- helpers
__device__ __forceinline__ float wred(float v) {
#pragma unroll
  for (int m = 16; m > 0; m >>= 1) v += __shfl_xor(v, m, 32);
  return v;
}
__device__ __forceinline__ float geluf(float x) {
  return 0.5f * x * (1.0f + erff(x * 0.70710678118654752440f));
}
// A fragment (16x32 bf16): lane l -> row l%16, K = {0..7,16..23} (l<16) or {8..15,24..31}
__device__ __forceinline__ v16bf load_a_frag(const bf16* p) {
  bf16x8 lo = *(const bf16x8*)(p);
  bf16x8 hi = *(const bf16x8*)(p + 16);
  v16bf a;
#pragma unroll
  for (int i = 0; i < 8; ++i) { a[i] = lo[i]; a[i + 8] = hi[i]; }
  return a;
}
// B fragment: pre-packed, lane reads its own 16 contiguous bf16
__device__ __forceinline__ v16bf load_b_frag(const bf16* p) {
  bf16x8 lo = *(const bf16x8*)(p);
  bf16x8 hi = *(const bf16x8*)(p + 8);
  v16bf b;
#pragma unroll
  for (int i = 0; i < 8; ++i) { b[i] = lo[i]; b[i + 8] = hi[i]; }
  return b;
}

// Row-token GEMM, register-blocked over MT m-tiles:
//   each wave owns whole 16-wide column strips (nt); every B fragment is loaded from
//   global exactly once per K-step and applied to MT LDS A fragments into MT
//   independent accumulators (no WMMA->WMMA RAW chain, 2-4x less L2 weight traffic).
// epi(row, col, val) is called once per output element owned by this lane.
template <int MT, typename EPI>
__device__ __forceinline__ void gemm_bf16(const bf16* __restrict__ Ab, int lda,
                                          const bf16* __restrict__ Bpk,
                                          int kTiles, int nTiles,
                                          int wave, int lane, EPI epi) {
  const int m16  = lane & 15;
  const int half = lane >> 4;
  const v8f zero = {0.f, 0.f, 0.f, 0.f, 0.f, 0.f, 0.f, 0.f};
  for (int nt = wave; nt < nTiles; nt += WAVES) {
    const bf16* bp = Bpk + (size_t)nt * kTiles * 512 + lane * 16;
    const bf16* ap = Ab + m16 * lda + half * 8;
    v8f acc[MT];
#pragma unroll
    for (int mt = 0; mt < MT; ++mt) acc[mt] = zero;
    for (int kt = 0; kt < kTiles; ++kt) {
      v16bf b = load_b_frag(bp + (size_t)kt * 512);
#pragma unroll
      for (int mt = 0; mt < MT; ++mt) {
        v16bf a = load_a_frag(ap + (mt * 16) * lda + kt * 32);
        acc[mt] = __builtin_amdgcn_wmma_f32_16x16x32_bf16(false, a, false, b, (short)0,
                                                          acc[mt], false, false);
      }
    }
    const int cb = nt * 16 + m16;
#pragma unroll
    for (int mt = 0; mt < MT; ++mt) {
      const int rb = mt * 16 + half * 8;
#pragma unroll
      for (int j = 0; j < 8; ++j) epi(rb + j, cb, acc[mt][j]);
    }
  }
}

// Per-row LayerNorm over 512 features: one wave per row, lane owns 16 contiguous cols.
__device__ __forceinline__ void ln_rows(const bf16* src, bf16* dst,
                                        const float* __restrict__ w,
                                        const float* __restrict__ b,
                                        int row0, int nrows, int wave, int lane) {
  for (int r = row0 + wave; r < row0 + nrows; r += WAVES) {
    const bf16* p = src + r * LDA + lane * 16;
    float x[16], s = 0.f, ss = 0.f;
#pragma unroll
    for (int i = 0; i < 16; ++i) { x[i] = (float)p[i]; s += x[i]; ss += x[i] * x[i]; }
    s = wred(s); ss = wred(ss);
    const float mean = s * (1.0f / 512.0f);
    const float var  = ss * (1.0f / 512.0f) - mean * mean;
    const float rstd = rsqrtf(var + 1e-5f);
    bf16* q = dst + r * LDA + lane * 16;
#pragma unroll
    for (int i = 0; i < 16; ++i) {
      const int f = lane * 16 + i;
      q[i] = (bf16)(((x[i] - mean) * rstd) * w[f] + b[f]);
    }
  }
}

// ---------------------------------------------------------------- weight repack (f32 -> bf16 WMMA-B tiles)
__global__ void pack_weight_kernel(const float* __restrict__ src, bf16* __restrict__ dst,
                                   int Ndim, int Kdim) {
  const int kTiles = Kdim >> 5;
  const int total  = Ndim * Kdim;
  for (int idx = blockIdx.x * blockDim.x + threadIdx.x; idx < total;
       idx += blockDim.x * gridDim.x) {
    const int i    = idx & 15;
    const int lane = (idx >> 4) & 31;
    const int t    = idx >> 9;           // fragment index = nt*kTiles + kt
    const int kt   = t % kTiles;
    const int nt   = t / kTiles;
    const int k = kt * 32 + ((i >> 3) << 4) + ((lane >> 4) << 3) + (i & 7);
    const int n = nt * 16 + (lane & 15);
    dst[idx] = (bf16)src[(size_t)n * Kdim + k];
  }
}

// ---------------------------------------------------------------- fused mixer
struct Smem {
  bf16  tk[ROWS * LDA];      // tokens (row = s*4+b), later residual stream of attention expert
  bf16  s1[ROWS * LDA];      // x1a / LN outputs
  bf16  fz[ROWS * LDA];      // fused accumulator  a0*x1 + a2*x3 + a1*x2
  bf16  s2[32 * LDS2];       // ctx / SE / qkv+ao / hidden / f32 output staging
  float sc[128];             // attention scores (8 samples x 16)
  float alpha[TILE_S * 3];   // gating weights
};

__global__ void __launch_bounds__(256, 1)
mixer_kernel(const float* __restrict__ branches,
             const float* __restrict__ conv_dw,
             const float* __restrict__ gn_w,  const float* __restrict__ gn_b,
             const float* __restrict__ ln_attn_w, const float* __restrict__ ln_attn_b,
             const float* __restrict__ in_proj_b, const float* __restrict__ attn_out_b,
             const float* __restrict__ ln_ffn_w,  const float* __restrict__ ln_ffn_b,
             const float* __restrict__ ffn_b1,    const float* __restrict__ ffn_b2,
             const float* __restrict__ mlp_ln_w,  const float* __restrict__ mlp_ln_b,
             const float* __restrict__ mlp_b1,    const float* __restrict__ mlp_b2,
             const float* __restrict__ gate_b1,   const float* __restrict__ gate_w2,
             const float* __restrict__ gate_b2,
             const bf16*  __restrict__ wpk,
             float* __restrict__ out) {
  __shared__ Smem sm;
  const int tid  = threadIdx.x;
  const int wave = tid >> 5, lane = tid & 31;
  const int n0   = blockIdx.x * TILE_S;

  // ---------- stage 0: load branches tile -> tk (bf16 tokens) + gating ctx (mean/max over b)
  for (int u = tid; u < TILE_S * 512; u += 256) {
    const int s = u >> 9, f = u & 511;
    const float4 v = *(const float4*)(branches + ((size_t)(n0 + s) * 512 + f) * 4);
    sm.tk[(s * 4 + 0) * LDA + f] = (bf16)v.x;
    sm.tk[(s * 4 + 1) * LDA + f] = (bf16)v.y;
    sm.tk[(s * 4 + 2) * LDA + f] = (bf16)v.z;
    sm.tk[(s * 4 + 3) * LDA + f] = (bf16)v.w;
    sm.s2[s * LDS2 + f]       = (bf16)(0.25f * (v.x + v.y + v.z + v.w));       // mean
    sm.s2[s * LDS2 + 512 + f] = (bf16)fmaxf(fmaxf(v.x, v.y), fmaxf(v.z, v.w)); // max
  }
  __syncthreads();

  // ---------- stage 1: gating  alpha = softmax(gelu(ctx@W1^T+b1)@W2^T+b2)
  gemm_bf16<1>(sm.s2, LDS2, wpk + OFF_GATE1, 32, 8, wave, lane,
               [&](int row, int col, float v) {
                 sm.s2[(16 + row) * LDS2 + col] = (bf16)geluf(v + gate_b1[col]);
               });
  __syncthreads();
  if (tid < TILE_S * 3) {
    const int s = tid / 3, e = tid % 3;
    float acc = gate_b2[e];
    for (int c = 0; c < 128; ++c)
      acc += (float)sm.s2[(16 + s) * LDS2 + c] * gate_w2[e * 128 + c];
    sm.alpha[tid] = acc;
  }
  __syncthreads();
  if (tid < TILE_S) {
    const float l0 = sm.alpha[tid * 3], l1 = sm.alpha[tid * 3 + 1], l2 = sm.alpha[tid * 3 + 2];
    const float m  = fmaxf(l0, fmaxf(l1, l2));
    const float e0 = expf(l0 - m), e1 = expf(l1 - m), e2 = expf(l2 - m);
    const float inv = 1.0f / (e0 + e1 + e2);
    sm.alpha[tid * 3] = e0 * inv; sm.alpha[tid * 3 + 1] = e1 * inv; sm.alpha[tid * 3 + 2] = e2 * inv;
  }
  __syncthreads();

  // ---------- stage 2: conv expert -> fz = alpha0 * gelu(GN(SE(dw(pw(x)))))
  gemm_bf16<4>(sm.tk, LDA, wpk + OFF_CONVPW, 16, 32, wave, lane,     // x1a -> s1
               [&](int row, int col, float v) { sm.s1[row * LDA + col] = (bf16)v; });
  __syncthreads();
  // depthwise (k=3 over branch axis, zero pad) + SE squeeze mean -> s2 rows0..15 cols512+
  for (int u = tid; u < TILE_S * 512; u += 256) {
    const int s = u >> 9, f = u & 511;
    const float w0 = conv_dw[f * 3 + 0], w1 = conv_dw[f * 3 + 1], w2 = conv_dw[f * 3 + 2];
    float acc = 0.f;
#pragma unroll
    for (int b = 0; b < 4; ++b) {
      const float xm = (b > 0) ? (float)sm.s1[(s * 4 + b - 1) * LDA + f] : 0.f;
      const float x0 = (float)sm.s1[(s * 4 + b) * LDA + f];
      const float xp = (b < 3) ? (float)sm.s1[(s * 4 + b + 1) * LDA + f] : 0.f;
      acc += w0 * xm + w1 * x0 + w2 * xp;
    }
    sm.s2[s * LDS2 + 512 + f] = (bf16)(0.25f * acc);
  }
  __syncthreads();
  gemm_bf16<1>(sm.s2 + 512, LDS2, wpk + OFF_SE1, 16, 4, wave, lane,  // SE fc1 + relu
               [&](int row, int col, float v) {
                 sm.s2[(16 + row) * LDS2 + 512 + col] = (bf16)fmaxf(v, 0.f);
               });
  __syncthreads();
  gemm_bf16<1>(sm.s2 + 16 * LDS2 + 512, LDS2, wpk + OFF_SE2, 2, 32, wave, lane, // fc2+sigmoid
               [&](int row, int col, float v) {
                 sm.s2[(16 + row) * LDS2 + col] = (bf16)(1.0f / (1.0f + expf(-v)));
               });
  __syncthreads();
  // scale, GroupNorm(1 group over F*B), gelu, weight by alpha0 -> fz   (one wave per sample)
  for (int s = wave; s < TILE_S; s += WAVES) {
    bf16* stash = sm.s2 + (wave * 4) * LDS2 + 512;   // private 4x512 stash per wave
    float sum = 0.f, ssq = 0.f;
    for (int e = lane; e < 2048; e += 32) {
      const int b = e >> 9, f = e & 511;
      const float w0 = conv_dw[f * 3 + 0], w1 = conv_dw[f * 3 + 1], w2 = conv_dw[f * 3 + 2];
      const float xm = (b > 0) ? (float)sm.s1[(s * 4 + b - 1) * LDA + f] : 0.f;
      const float x0 = (float)sm.s1[(s * 4 + b) * LDA + f];
      const float xp = (b < 3) ? (float)sm.s1[(s * 4 + b + 1) * LDA + f] : 0.f;
      const float x  = (w0 * xm + w1 * x0 + w2 * xp) * (float)sm.s2[(16 + s) * LDS2 + f];
      stash[b * LDS2 + f] = (bf16)x;
      sum += x; ssq += x * x;
    }
    sum = wred(sum); ssq = wred(ssq);
    const float mean = sum * (1.0f / 2048.0f);
    const float var  = ssq * (1.0f / 2048.0f) - mean * mean;
    const float rstd = rsqrtf(var + 1e-5f);
    const float a0   = sm.alpha[s * 3 + 0];
    for (int e = lane; e < 2048; e += 32) {
      const int b = e >> 9, f = e & 511;
      const float x = (float)stash[b * LDS2 + f];
      const float y = (x - mean) * rstd * gn_w[f] + gn_b[f];
      sm.fz[(s * 4 + b) * LDA + f] = (bf16)(a0 * geluf(y));
    }
  }
  __syncthreads();

  // ---------- stage 3: MLP expert -> fz += alpha2 * (t + gelu(LN(t)@W1^T+b1)@W2^T+b2)
  ln_rows(sm.tk, sm.s1, mlp_ln_w, mlp_ln_b, 0, ROWS, wave, lane);
  __syncthreads();
  for (int h = 0; h < 2; ++h) {            // 32-row halves (hidden buffer is 32x1024)
    gemm_bf16<2>(sm.s1 + h * 32 * LDA, LDA, wpk + OFF_MLP1, 16, 64, wave, lane,
                 [&](int row, int col, float v) {
                   sm.s2[row * LDS2 + col] = (bf16)geluf(v + mlp_b1[col]);
                 });
    __syncthreads();
    gemm_bf16<2>(sm.s2, LDS2, wpk + OFF_MLP2, 32, 32, wave, lane,
                 [&](int row, int col, float v) {
                   const int gr = h * 32 + row;
                   const float x3 = (float)sm.tk[gr * LDA + col] + v + mlp_b2[col];
                   const float a2 = sm.alpha[(gr >> 2) * 3 + 2];
                   sm.fz[gr * LDA + col] = (bf16)((float)sm.fz[gr * LDA + col] + a2 * x3);
                 });
    __syncthreads();
  }

  // ---------- stage 4: attention expert (per-sample, 4 tokens, 4 heads of dh=128), 32-row halves
  for (int hh = 0; hh < 2; ++hh) {
    const int g0 = hh * 32;
    ln_rows(sm.tk, sm.s1, ln_attn_w, ln_attn_b, g0, 32, wave, lane);
    __syncthreads();
    for (int hd = 0; hd < 4; ++hd) {
#pragma unroll
      for (int p = 0; p < 3; ++p) {        // q / k / v projections for this head
        const int   nbase = p * 512 + hd * 128;
        const float qs    = (p == 0) ? 0.08838834764831845f : 1.0f;  // dh^-0.5 on q
        gemm_bf16<2>(sm.s1 + g0 * LDA, LDA,
                     wpk + OFF_INPROJ + (size_t)(nbase >> 4) * (16 * 512), 16, 8, wave, lane,
                     [&](int row, int col, float v) {
                       sm.s2[row * LDS2 + p * 128 + col] =
                           (bf16)((v + in_proj_b[nbase + col]) * qs);
                     });
      }
      __syncthreads();
      {  // scores + softmax + ao: one wave per local sample
        const int p = lane;
        float scv = 0.f;
        if (p < 16) {
          const int qb = p >> 2, kb = p & 3;
          const bf16* qr = sm.s2 + (wave * 4 + qb) * LDS2;
          const bf16* kr = sm.s2 + (wave * 4 + kb) * LDS2 + 128;
          for (int d = 0; d < 128; ++d) scv += (float)qr[d] * (float)kr[d];
        }
        float m = fmaxf(scv, __shfl_xor(scv, 1, 32));
        m = fmaxf(m, __shfl_xor(m, 2, 32));
        float ev = expf(scv - m);
        float es = ev + __shfl_xor(ev, 1, 32);
        es += __shfl_xor(es, 2, 32);
        if (p < 16) sm.sc[wave * 16 + p] = ev / es;
        for (int ii = lane; ii < 512; ii += 32) {   // ao = att @ v  -> s2 cols 512+hd*128
          const int qb = ii >> 7, d = ii & 127;
          float a = 0.f;
#pragma unroll
          for (int kb = 0; kb < 4; ++kb)
            a += sm.sc[wave * 16 + qb * 4 + kb] * (float)sm.s2[(wave * 4 + kb) * LDS2 + 256 + d];
          sm.s2[(wave * 4 + qb) * LDS2 + 512 + hd * 128 + d] = (bf16)a;
        }
      }
      __syncthreads();
    }
    gemm_bf16<2>(sm.s2 + 512, LDS2, wpk + OFF_ATTNOUT, 16, 32, wave, lane,  // out proj + residual
                 [&](int row, int col, float v) {
                   const int gr = g0 + row;
                   sm.tk[gr * LDA + col] =
                       (bf16)((float)sm.tk[gr * LDA + col] + v + attn_out_b[col]);
                 });
    __syncthreads();
    ln_rows(sm.tk, sm.s1, ln_ffn_w, ln_ffn_b, g0, 32, wave, lane);
    __syncthreads();
    gemm_bf16<2>(sm.s1 + g0 * LDA, LDA, wpk + OFF_FFN1, 16, 64, wave, lane,
                 [&](int row, int col, float v) {
                   sm.s2[row * LDS2 + col] = (bf16)geluf(v + ffn_b1[col]);
                 });
    __syncthreads();
    gemm_bf16<2>(sm.s2, LDS2, wpk + OFF_FFN2, 32, 32, wave, lane,
                 [&](int row, int col, float v) {
                   const int gr = g0 + row;
                   const float x2 = (float)sm.tk[gr * LDA + col] + v + ffn_b2[col];
                   const float a1 = sm.alpha[(gr >> 2) * 3 + 1];
                   sm.fz[gr * LDA + col] = (bf16)((float)sm.fz[gr * LDA + col] + a1 * x2);
                 });
    __syncthreads();
  }

  // ---------- stage 5: out = fz @ outp_w^T, staged through LDS f32 for coalesced float4 stores
  float* s2f = (float*)sm.s2;
  for (int hh = 0; hh < 2; ++hh) {
    gemm_bf16<2>(sm.fz + hh * 32 * LDA, LDA, wpk + OFF_OUTP, 16, 32, wave, lane,
                 [&](int row, int col, float v) { s2f[row * LDS2F + col] = v; });
    __syncthreads();
    for (int u = tid; u < 8 * 512; u += 256) {
      const int ls = u >> 9, o = u & 511;
      float4 r;
      r.x = s2f[(ls * 4 + 0) * LDS2F + o];
      r.y = s2f[(ls * 4 + 1) * LDS2F + o];
      r.z = s2f[(ls * 4 + 2) * LDS2F + o];
      r.w = s2f[(ls * 4 + 3) * LDS2F + o];
      *(float4*)(out + ((size_t)(n0 + hh * 8 + ls) * 512 + o) * 4) = r;
    }
    __syncthreads();
  }
}

// ---------------------------------------------------------------- launch
extern "C" void kernel_launch(void* const* d_in, const int* in_sizes, int n_in,
                              void* d_out, int out_size, void* d_ws, size_t ws_size,
                              hipStream_t stream) {
  bf16* wpk = (bf16*)d_ws;   // ~7.4 MB of packed bf16 weights
  struct PW { int idx; unsigned off; int N, K; };
  const PW pws[11] = {
      {1,  OFF_CONVPW,  512,  512}, {9,  OFF_INPROJ, 1536, 512},
      {11, OFF_ATTNOUT, 512,  512}, {15, OFF_FFN1,   1024, 512},
      {17, OFF_FFN2,    512, 1024}, {21, OFF_MLP1,   1024, 512},
      {23, OFF_MLP2,    512, 1024}, {25, OFF_GATE1,  128, 1024},
      {5,  OFF_SE1,      64,  512}, {6,  OFF_SE2,    512,   64},
      {29, OFF_OUTP,    512,  512}};
  for (int i = 0; i < 11; ++i) {
    const int total = pws[i].N * pws[i].K;
    pack_weight_kernel<<<(total + 255) / 256, 256, 0, stream>>>(
        (const float*)d_in[pws[i].idx], wpk + pws[i].off, pws[i].N, pws[i].K);
  }
  mixer_kernel<<<16384 / TILE_S, 256, 0, stream>>>(
      (const float*)d_in[0],  (const float*)d_in[2],  (const float*)d_in[3],
      (const float*)d_in[4],  (const float*)d_in[7],  (const float*)d_in[8],
      (const float*)d_in[10], (const float*)d_in[12], (const float*)d_in[13],
      (const float*)d_in[14], (const float*)d_in[16], (const float*)d_in[18],
      (const float*)d_in[19], (const float*)d_in[20], (const float*)d_in[22],
      (const float*)d_in[24], (const float*)d_in[26], (const float*)d_in[27],
      (const float*)d_in[28], (const bf16*)d_ws, (float*)d_out);
}
